// MCGRUModel_69037304316199
// MI455X (gfx1250) — compile-verified, matrix operand order checked by reading, not run on hardware
//
#include <hip/hip_runtime.h>
#include <hip/hip_bf16.h>
#include <stdint.h>

// Problem constants (match reference)
#define BB   256   // batch
#define TT   256   // timesteps
#define DD   128   // lab channels
#define HH   8     // per-channel GRU hidden
#define G3   24    // 3*H
#define HID  32
#define SD   32

typedef __attribute__((ext_vector_type(16))) _Float16 v16h;
typedef __attribute__((ext_vector_type(8)))  float    v8f;
typedef unsigned __attribute__((ext_vector_type(4))) uint4v;
typedef int      __attribute__((ext_vector_type(8))) int8v;
typedef int      __attribute__((ext_vector_type(4))) int4v;

// ---------------------------------------------------------------------------
// Kernel 1: xl = x @ lab_W + lab_b   via v_wmma_f32_16x16x32_f16.
// The 16x128 f32 activation tile is staged global->LDS with the Tensor Data
// Mover (tensor_load_to_lds, TENSORcnt), then packed to f16 pairs in LDS.
// M = B*T = 65536 rows, N = K = 128. One block = one 16-row M tile, 8 waves
// cover the 8 N tiles. K loop: 4 iterations of K=32.
// ---------------------------------------------------------------------------
__global__ __launch_bounds__(256) void lab_gemm_wmma(
    const float* __restrict__ x, const float* __restrict__ labW,
    const float* __restrict__ labb, float* __restrict__ xl)
{
    __shared__ float    sxf[16 * 128];  // raw f32 tile from TDM  (8 KB)
    __shared__ unsigned sx[16 * 64];    // [m16][pair] packed f16 (4 KB)
    __shared__ unsigned sw[128 * 64];   // [n][pair]   packed f16 (32 KB)

    const int tid   = threadIdx.x;
    const int mbase = blockIdx.x * 16;

    // --- TDM: DMA the 16x128 f32 x-tile into LDS (one wave issues it) -----
    if (tid < 32) {
        const uint64_t gaddr = (uint64_t)(uintptr_t)(x + (size_t)mbase * DD);
        const unsigned ldsa  = (unsigned)(size_t)(&sxf[0]);
        // D# group 0: count=1 | lds_addr | global_addr[56:0] | type=2
        uint4v g0 = { 1u,
                      ldsa,
                      (unsigned)(gaddr & 0xFFFFFFFFu),
                      (unsigned)((gaddr >> 32) & 0x1FFFFFFu) | (2u << 30) };
        // D# group 1: data_size=4B; tensor 128 x 16; tile 128 x 16; stride0=128
        int8v g1 = { (int)(2u << 16),   // workgroup_mask=0, data_size=2 (4B)
                     (int)(128u << 16), // tensor_dim0[15:0]=128
                     (int)(16u  << 16), // tensor_dim0[31:16]=0, tensor_dim1 lo=16
                     (int)(128u << 16), // tensor_dim1 hi=0, tile_dim0=128
                     16,                // tile_dim1=16, tile_dim2=0
                     128,               // tensor_dim0_stride lo32 = 128 elements
                     0, 0 };
        int4v g2 = { 0, 0, 0, 0 };      // unused dims (2-D tensor)
        int4v g3 = { 0, 0, 0, 0 };
        int8v g4 = { 0, 0, 0, 0, 0, 0, 0, 0 };   // trailing group (unused)
        __builtin_amdgcn_tensor_load_to_lds(g0, g1, g2, g3, g4, 0);
        __builtin_amdgcn_s_wait_tensorcnt(0);
    }

    // Stage lab_W (all 128 columns), n fastest for coalescing (L2-resident).
    for (int idx = tid; idx < 128 * 64; idx += 256) {
        const int n = idx & 127, p = idx >> 7;
        union { _Float16 h[2]; unsigned u; } pk;
        pk.h[0] = (_Float16)labW[(2 * p)     * DD + n];
        pk.h[1] = (_Float16)labW[(2 * p + 1) * DD + n];
        sw[n * 64 + p] = pk.u;
    }
    __syncthreads();

    // Pack the DMA'd f32 tile into f16 pairs laid out for the A fragment.
    for (int idx = tid; idx < 16 * 64; idx += 256) {
        const int m16 = idx >> 6, p = idx & 63;
        union { _Float16 h[2]; unsigned u; } pk;
        pk.h[0] = (_Float16)sxf[m16 * 128 + 2 * p];
        pk.h[1] = (_Float16)sxf[m16 * 128 + 2 * p + 1];
        sx[idx] = pk.u;
    }
    __syncthreads();

    const int lane = tid & 31;
    const int wave = tid >> 5;        // n-tile id (0..7)
    const int hi   = lane >> 4;       // K-half select per A/B striping
    const int l15  = lane & 15;
    const int ncol = wave * 16 + l15;

    v8f c = {};
    union { v16h v; unsigned u[8]; } a, b;

    #pragma unroll
    for (int kit = 0; kit < 4; ++kit) {
        const int kb = kit * 16;      // pair base for this K=32 chunk
        #pragma unroll
        for (int v = 0; v < 4; ++v) {
            const int p0 = kb + v + 4 * hi;        // K = {0..7}  (+8 for hi)
            const int p1 = kb + 8 + v + 4 * hi;    // K = {16..23}(+8 for hi)
            a.u[v]     = sx[l15  * 64 + p0];
            a.u[v + 4] = sx[l15  * 64 + p1];
            b.u[v]     = sw[ncol * 64 + p0];
            b.u[v + 4] = sw[ncol * 64 + p1];
        }
        c = __builtin_amdgcn_wmma_f32_16x16x32_f16(
                /*neg_a=*/false, a.v, /*neg_b=*/false, b.v,
                /*c_mod=*/(short)0, c, /*reuse_a=*/false, /*reuse_b=*/false);
    }

    const float bias = labb[ncol];
    #pragma unroll
    for (int j = 0; j < 8; ++j) {
        const int m = mbase + j + 8 * hi;          // C layout: VGPR j -> M=j(+8)
        xl[(size_t)m * DD + ncol] = c[j] + bias;
    }
}

// ---------------------------------------------------------------------------
// Kernel 2: fused per-channel GRU scan. Thread = one (b,d) cell; h[8] lives in
// registers. W_hh for a 32-channel slice in LDS; W_ih / (b_ih+b_hh) in regs.
// Snapshot h at t == lengths[b]-1 (only state the reference output needs).
// Block = 256 threads = 32 d  x 8 b; grid = 4 d-groups x 32 b-groups = 128.
// ---------------------------------------------------------------------------
__global__ __launch_bounds__(256) void mcgru_scan(
    const float* __restrict__ xl, const int* __restrict__ lengths,
    const float* __restrict__ Wih, const float* __restrict__ Whh,
    const float* __restrict__ bih, const float* __restrict__ bhh,
    float* __restrict__ hsel)
{
    __shared__ float sW[32 * 192];   // W_hh slice: 24 KB

    const int tid   = threadIdx.x;
    const int dl    = tid & 31;
    const int bl    = tid >> 5;
    const int dbase = (blockIdx.x & 3) * 32;
    const int bg    = (blockIdx.x >> 2) * 8 + bl;
    const int d     = dbase + dl;

    for (int idx = tid; idx < 32 * 192; idx += 256)
        sW[idx] = Whh[(size_t)dbase * 192 + idx];
    __syncthreads();

    float wih[G3], bsum[G3];
    #pragma unroll
    for (int g = 0; g < G3; ++g) {
        wih[g]  = Wih[d * G3 + g];
        bsum[g] = bih[d * G3 + g] + bhh[d * G3 + g];
    }

    float h[HH];
    #pragma unroll
    for (int i = 0; i < HH; ++i) h[i] = 0.0f;

    const int lenb = lengths[bg];
    const float* __restrict__ sWd  = &sW[dl * 192];
    const float* __restrict__ xrow = xl + (size_t)bg * TT * DD + d;

    for (int t = 0; t < TT; ++t) {
        const float xv = xrow[(size_t)t * DD];
        if (t + 16 < TT)
            __builtin_prefetch(xrow + (size_t)(t + 16) * DD, 0, 1);

        float r_[HH], z_[HH], hn[HH];
        #pragma unroll
        for (int i = 0; i < HH; ++i) {           // reset gate
            float acc = bsum[i] + xv * wih[i];
            #pragma unroll
            for (int k = 0; k < HH; ++k) acc += h[k] * sWd[i * 8 + k];
            r_[i] = 1.0f / (1.0f + __expf(-acc));
        }
        #pragma unroll
        for (int i = 0; i < HH; ++i) {           // update gate
            float acc = bsum[8 + i] + xv * wih[8 + i];
            #pragma unroll
            for (int k = 0; k < HH; ++k) acc += h[k] * sWd[(8 + i) * 8 + k];
            z_[i] = 1.0f / (1.0f + __expf(-acc));
        }
        #pragma unroll
        for (int i = 0; i < HH; ++i) {           // candidate + blend
            float acc = 0.0f;
            #pragma unroll
            for (int k = 0; k < HH; ++k) acc += h[k] * sWd[(16 + i) * 8 + k];
            const float nv = bsum[16 + i] + xv * wih[16 + i] + r_[i] * acc;
            const float e  = __expf(-2.0f * nv);
            const float th = (1.0f - e) / (1.0f + e);
            hn[i] = (1.0f - z_[i]) * th + z_[i] * h[i];
        }
        #pragma unroll
        for (int i = 0; i < HH; ++i) h[i] = hn[i];

        if (t == lenb - 1) {
            #pragma unroll
            for (int i = 0; i < HH; ++i)
                hsel[(size_t)bg * (DD * HH) + d * HH + i] = h[i];
        }
    }
}

// ---------------------------------------------------------------------------
// Kernel 3: demo = static@static_W+static_b; h_out = [demo|h_last]@out_W+out_b
// (only at the gathered timestep); y = h_out@head_W+head_b. One wave per b.
// ---------------------------------------------------------------------------
__global__ __launch_bounds__(32) void head_kernel(
    const float* __restrict__ statc, const float* __restrict__ staticW,
    const float* __restrict__ staticb, const float* __restrict__ hsel,
    const float* __restrict__ outW, const float* __restrict__ outb,
    const float* __restrict__ headW, const float* __restrict__ headb,
    float* __restrict__ out)
{
    __shared__ float demo[HID];
    __shared__ float hout[HID];
    const int b = blockIdx.x;
    const int n = threadIdx.x;

    float dacc = staticb[n];
    #pragma unroll 4
    for (int s = 0; s < SD; ++s) dacc += statc[b * SD + s] * staticW[s * HID + n];
    demo[n] = dacc;
    __syncthreads();

    float acc = outb[n];
    #pragma unroll 4
    for (int j = 0; j < HID; ++j) acc += demo[j] * outW[j * HID + n];
    const float* hb = hsel + (size_t)b * (DD * HH);
    for (int k = 0; k < DD * HH; ++k) acc += hb[k] * outW[(HID + k) * HID + n];
    hout[n] = acc;
    __syncthreads();

    if (n == 0) {
        float y = headb[0];
        #pragma unroll 4
        for (int j = 0; j < HID; ++j) y += hout[j] * headW[j];
        out[b] = y;
    }
}

// ---------------------------------------------------------------------------
extern "C" void kernel_launch(void* const* d_in, const int* in_sizes, int n_in,
                              void* d_out, int out_size, void* d_ws, size_t ws_size,
                              hipStream_t stream)
{
    const float* x        = (const float*)d_in[0];
    const int*   lengths  = (const int*)  d_in[1];
    const float* statc    = (const float*)d_in[2];
    const float* static_W = (const float*)d_in[3];
    const float* static_b = (const float*)d_in[4];
    const float* lab_W    = (const float*)d_in[5];
    const float* lab_b    = (const float*)d_in[6];
    const float* W_ih     = (const float*)d_in[7];
    const float* W_hh     = (const float*)d_in[8];
    const float* b_ih     = (const float*)d_in[9];
    const float* b_hh     = (const float*)d_in[10];
    const float* out_W    = (const float*)d_in[11];
    const float* out_b    = (const float*)d_in[12];
    const float* head_W   = (const float*)d_in[13];
    const float* head_b   = (const float*)d_in[14];

    float* xl   = (float*)d_ws;                                     // 32 MB
    float* hsel = (float*)((char*)d_ws + (size_t)BB * TT * DD * 4); // +1 MB
    float* out  = (float*)d_out;

    lab_gemm_wmma<<<(BB * TT) / 16, 256, 0, stream>>>(x, lab_W, lab_b, xl);
    mcgru_scan<<<(BB / 8) * (DD / 32), 256, 0, stream>>>(
        xl, lengths, W_ih, W_hh, b_ih, b_hh, hsel);
    head_kernel<<<BB, 32, 0, stream>>>(
        statc, static_W, static_b, hsel, out_W, out_b, head_W, head_b, out);
}
